// DiffeqZeroTraceAttention_42932493091574
// MI455X (gfx1250) — matrix operand order repaired and checked by reading, not compiled, not run on hardware
//
#include <hip/hip_runtime.h>
#include <hip/hip_bf16.h>

typedef __attribute__((ext_vector_type(16))) _Float16 v16h;
typedef __attribute__((ext_vector_type(8)))  _Float16 v8h;
typedef __attribute__((ext_vector_type(8)))  float    v8f;
typedef unsigned int u32x4 __attribute__((ext_vector_type(4)));
typedef int          i32x8 __attribute__((ext_vector_type(8)));
typedef int          i32x4 __attribute__((ext_vector_type(4)));

#define NB 8
#define NN 1024
#define RR (NB*NN)      // 8192 rows (B*N)
#define H1C 256
#define H2C 256
#define HDC 128
#define NHC 4
#define DHC 32

// ---------------------------------------------------------------------------
// TDM helper: async-copy one contiguous 8 KB row-panel (16 rows x 256 f16)
// from global memory into LDS via the Tensor Data Mover.
// D# packing per CDNA5 ISA ch.8: group0 = {flags, lds_addr, global_addr, type},
// group1 = {data_size, tensor dims/strides, tile dims} (4-byte elements,
// tile = 1 x 2048 dwords). Tracked by TENSORcnt.
// ---------------------------------------------------------------------------
__device__ __forceinline__ void tdm_load_panel_8k(const void* gsrc, void* lds_dst) {
  unsigned long long ga = (unsigned long long)(uintptr_t)gsrc;        // byte addr
  unsigned int la = (unsigned int)(uintptr_t)lds_dst;                 // LDS offset = addr[31:0]
  u32x4 g0;
  g0[0] = 1u;                                  // count=1, user descriptor
  g0[1] = la;                                  // lds_addr (bits 63:32)
  g0[2] = (unsigned int)ga;                    // global_addr low  (bits 95:64)
  g0[3] = (unsigned int)(ga >> 32) | (2u << 30); // global_addr high | type=2 (127:126)
  i32x8 g1;
  g1[0] = (2 << 16);                           // workgroup_mask=0, data_size=2 (4B)
  g1[1] = (int)(2048u << 16);                  // tensor_dim0[15:0]=2048 (bits 63:48)
  g1[2] = (int)(1u << 16);                     // tensor_dim0[31:16]=0, tensor_dim1[15:0]=1
  g1[3] = (int)(2048u << 16);                  // tensor_dim1 hi=0, tile_dim0=2048 (127:112)
  g1[4] = 1;                                   // tile_dim1=1, tile_dim2=0
  g1[5] = 2048;                                // tensor_dim0_stride low
  g1[6] = 0;                                   // stride hi / dim1_stride low
  g1[7] = 0;
  i32x4 z4 = {0, 0, 0, 0};
#if __has_include(<hip/amd_detail/amd_gfx1250_TDM.h>)
  i32x8 z8 = {0, 0, 0, 0, 0, 0, 0, 0};
  __builtin_amdgcn_tensor_load_to_lds(g0, g1, z4, z4, z8, 0);   // 6-arg toolchain
#else
  __builtin_amdgcn_tensor_load_to_lds(g0, g1, z4, z4, 0);       // 5-arg ROCm 7.2
#endif
}

// ---------------------------------------------------------------------------
// Weight conversion: fp32 -> f16, folding MADE masks (deg = idx%3 + 1).
// maskmode 0: none; 1: M2 keep (r%3)>=(c%3); 2: M3 keep (r>>7)>(c%3).
// ---------------------------------------------------------------------------
__global__ void convert_w_kernel(const float* __restrict__ w, _Float16* __restrict__ o,
                                 int rows, int cols, int maskmode) {
  int idx = blockIdx.x * blockDim.x + threadIdx.x;
  if (idx >= rows * cols) return;
  int r = idx / cols, c = idx % cols;
  float val = w[idx];
  if (maskmode == 1) { if ((r % 3) < (c % 3)) val = 0.f; }
  else if (maskmode == 2) { if ((r >> 7) <= (c % 3)) val = 0.f; }
  o[idx] = (_Float16)val;
}

// ---------------------------------------------------------------------------
// Layer 1 (K=4, trivial): q masked (M1: allow d <= h%3) + relu; k,v full + tanh.
// ---------------------------------------------------------------------------
__global__ void layer1_kernel(const float* __restrict__ x,
                              const float* __restrict__ qw1, const float* __restrict__ qb1,
                              const float* __restrict__ kw1, const float* __restrict__ kb1,
                              const float* __restrict__ vw1, const float* __restrict__ vb1,
                              _Float16* __restrict__ h1q, _Float16* __restrict__ h1k,
                              _Float16* __restrict__ h1v) {
  int idx = blockIdx.x * blockDim.x + threadIdx.x;   // over RR*256
  int row = idx >> 8, hc = idx & 255;
  float x0 = x[row*4+0], x1 = x[row*4+1], x2 = x[row*4+2], x3 = x[row*4+3];
  int lim = hc % 3;                                  // M1: d <= lim (d=3 never allowed)
  const float* wq = qw1 + hc*4;
  float aq = qb1[hc] + wq[0]*x0;
  if (lim >= 1) aq += wq[1]*x1;
  if (lim >= 2) aq += wq[2]*x2;
  h1q[idx] = (_Float16)fmaxf(aq, 0.f);
  const float* wk = kw1 + hc*4;
  h1k[idx] = (_Float16)tanhf(kb1[hc] + wk[0]*x0 + wk[1]*x1 + wk[2]*x2 + wk[3]*x3);
  const float* wv = vw1 + hc*4;
  h1v[idx] = (_Float16)tanhf(vb1[hc] + wv[0]*x0 + wv[1]*x1 + wv[2]*x2 + wv[3]*x3);
}

// ---------------------------------------------------------------------------
// WMMA GEMM with TDM-staged activations:
//   out = act(A[RR,256] @ W[Cout,256]^T + bias)
// Block = 1 row-tile x 8 col-tiles (8 waves). The 16x256 A panel (8 KB) is
// shared by all 8 waves: one TDM async copy into LDS (TENSORcnt wait +
// barrier), then per-k-step ds_load_b128 A-fragments while W streams from
// global (W is <=256 KB -> L2/WGP$ resident across all 512 row-tiles).
// K=256 -> 8 chained v_wmma_f32_16x16x32_f16.
// act: 0=relu 1=tanh 2=none.  storeMode: 0 -> [RR,Cout] f16,
// 1 -> transposed V layout [(b*128 + c), n] for the attention PV fragments.
// ---------------------------------------------------------------------------
__global__ void gemm_k256_kernel(const _Float16* __restrict__ A,
                                 const _Float16* __restrict__ W,
                                 const float* __restrict__ bias,
                                 _Float16* __restrict__ out,
                                 int Cout, int act, int storeMode) {
  __shared__ __align__(128) _Float16 sA[16 * 256];   // 8 KB staged row panel
  int wave = threadIdx.x >> 5, lane = threadIdx.x & 31;
  int blocksPerRow = Cout >> 7;                      // Cout/128
  int rowTile = blockIdx.x / blocksPerRow;
  int colTile = (blockIdx.x - rowTile * blocksPerRow) * 8 + wave;
  int m = lane & 15, grp = lane >> 4;

  if (wave == 0) {                                   // one TDM issue per block
    tdm_load_panel_8k(A + (size_t)rowTile * 16 * 256, sA);
    __builtin_amdgcn_s_wait_tensorcnt(0);
  }
  __syncthreads();

  const _Float16* arow = &sA[m * 256];               // A-frag: lane = M row (LDS)
  int c = colTile * 16 + m;                          // B-frag: lane = N col
  const _Float16* wrow = W + (size_t)c * 256;
  v8f acc = {};
#pragma unroll
  for (int kb = 0; kb < 256; kb += 32) {
    union { v16h v; v8h h2[2]; } a;
    a.h2[0] = *(const v8h*)(arow + kb + grp*16);     // K 0-7  | 16-23
    a.h2[1] = *(const v8h*)(arow + kb + grp*16 + 8); // K 8-15 | 24-31
    v16h b = *(const v16h*)(wrow + kb + grp*16);     // K = grp*16 + i
    acc = __builtin_amdgcn_wmma_f32_16x16x32_f16(false, a.v, false, b,
                                                 (short)0, acc, false, false);
  }
  float bv = bias[c];
#pragma unroll
  for (int v = 0; v < 8; ++v) {
    int rr = rowTile*16 + v + 8*grp;                 // D-frag: VGPR v -> M row
    float o = acc[v] + bv;
    if (act == 0) o = fmaxf(o, 0.f);
    else if (act == 1) o = tanhf(o);
    if (storeMode == 0) {
      out[(size_t)rr * Cout + c] = (_Float16)o;
    } else {
      int b_ = rr >> 10, n = rr & 1023;              // c = h*32 + e
      out[(size_t)(b_*HDC + c) * NN + n] = (_Float16)o;
    }
  }
}

// ---------------------------------------------------------------------------
// Flash attention with zero-trace (diag = -inf). One wave per (b,d,h,qtile16).
// Computes S^T = K_tile x Q (lane owns one query column -> cheap row softmax),
// then O^T = V^T x P^T.  32 keys per iteration, online max/sum rescaling.
// ---------------------------------------------------------------------------
__global__ void attention_kernel(const _Float16* __restrict__ Q,   // [RR,512]
                                 const _Float16* __restrict__ K,   // [RR,128]
                                 const _Float16* __restrict__ VT,  // [NB*128,1024]
                                 float* __restrict__ Y) {          // [NB,NN,4,128] f32
  int wave = threadIdx.x >> 5, lane = threadIdx.x & 31;
  int tid = blockIdx.x * 8 + wave;
  int qt = tid & 63; int rest = tid >> 6;
  int h = rest & 3; rest >>= 2;
  int d = rest & 3; int b = rest >> 2;
  int m = lane & 15, grp = lane >> 4;
  int qabs = qt*16 + m;

  // Q as B-fragment: lane column = query, halves = e (grp*16 + i). One 32B load.
  v16h qf = *(const v16h*)(Q + ((size_t)(b*NN + qabs))*512 + d*HDC + h*DHC + grp*16);

  v8f acc0 = {}, acc1 = {};
  float mrun = -3.0e38f, lrun = 0.f;
  const float scale = 0.1767766952966369f;             // 1/sqrt(32)

  for (int kb = 0; kb < NN; kb += 32) {
    const _Float16* k1p = K + ((size_t)(b*NN + kb + m))*HDC + h*DHC;
    const _Float16* k2p = K + ((size_t)(b*NN + kb + 16 + m))*HDC + h*DHC;
    if (kb + 32 < NN) __builtin_prefetch(k1p + 32*HDC, 0, 1);   // next key tile
    union { v16h v; v8h h2[2]; } ka, kc;
    ka.h2[0] = *(const v8h*)(k1p + grp*16);
    ka.h2[1] = *(const v8h*)(k1p + grp*16 + 8);
    kc.h2[0] = *(const v8h*)(k2p + grp*16);
    kc.h2[1] = *(const v8h*)(k2p + grp*16 + 8);
    v8f s1 = {}, s2 = {};
    s1 = __builtin_amdgcn_wmma_f32_16x16x32_f16(false, ka.v, false, qf, (short)0, s1, false, false);
    s2 = __builtin_amdgcn_wmma_f32_16x16x32_f16(false, kc.v, false, qf, (short)0, s2, false, false);

    float p1[8], p2[8], tm = -3.0e38f;
#pragma unroll
    for (int v = 0; v < 8; ++v) {
      int ki1 = kb + v + 8*grp;                        // S^T rows = keys
      int ki2 = kb + 16 + v + 8*grp;
      float a1 = s1[v]*scale; if (ki1 == qabs) a1 = -3.0e38f;   // zero-trace diag
      float a2 = s2[v]*scale; if (ki2 == qabs) a2 = -3.0e38f;
      p1[v] = a1; p2[v] = a2;
      tm = fmaxf(tm, fmaxf(a1, a2));
    }
    tm = fmaxf(tm, __shfl_xor(tm, 16, 32));            // lane L and L^16 share a query
    float mnew = fmaxf(mrun, tm);
    float alpha = __expf(mrun - mnew);
    float tsum = 0.f;
#pragma unroll
    for (int v = 0; v < 8; ++v) {
      p1[v] = __expf(p1[v] - mnew);
      p2[v] = __expf(p2[v] - mnew);
      tsum += p1[v] + p2[v];
    }
    tsum += __shfl_xor(tsum, 16, 32);
    lrun = lrun * alpha + tsum;
    mrun = mnew;

    // Repack probabilities into a B-fragment (lane=query col, k = 8*grp+i | 16+8*grp+i).
    v16h pf;
#pragma unroll
    for (int v = 0; v < 8; ++v) {
      float e1 = __shfl_xor(p1[v], 16, 32);
      float e2 = __shfl_xor(p2[v], 16, 32);
      pf[v]   = (_Float16)p1[v];
      pf[v+8] = (_Float16)p2[v];
      acc0[v] *= alpha;
      acc1[v] *= alpha;
    }
    // V^T A-fragments: rows = e (from pre-transposed VT), K = keys. 16B loads.
    const _Float16* v0p = VT + ((size_t)(b*HDC + h*DHC + m)) * (size_t)NN;
    const _Float16* v1p = VT + ((size_t)(b*HDC + h*DHC + 16 + m)) * (size_t)NN;
    union { v16h v; v8h h2[2]; } va, vb;
    va.h2[0] = *(const v8h*)(v0p + kb + grp*8);
    va.h2[1] = *(const v8h*)(v0p + kb + 16 + grp*8);
    vb.h2[0] = *(const v8h*)(v1p + kb + grp*8);
    vb.h2[1] = *(const v8h*)(v1p + kb + 16 + grp*8);
    acc0 = __builtin_amdgcn_wmma_f32_16x16x32_f16(false, va.v, false, pf, (short)0, acc0, false, false);
    acc1 = __builtin_amdgcn_wmma_f32_16x16x32_f16(false, vb.v, false, pf, (short)0, acc1, false, false);
  }
  float inv = 1.f / lrun;
  size_t ybase = (((size_t)(b*NN + qabs))*4 + d) * HDC + h*DHC;
#pragma unroll
  for (int v = 0; v < 8; ++v) {
    int e0 = v + 8*grp;
    Y[ybase + e0]      = acc0[v] * inv;
    Y[ybase + 16 + e0] = acc1[v] * inv;
  }
}

// ---------------------------------------------------------------------------
// Final projection (HD->1 per dim) + zero log-det output.
// ---------------------------------------------------------------------------
__global__ void proj_kernel(const float* __restrict__ Y, const float* __restrict__ pw,
                            const float* __restrict__ pb, float* __restrict__ out) {
  int idx = blockIdx.x * blockDim.x + threadIdx.x;     // over RR*4 (b,n,d)
  if (idx >= RR*4) return;
  const float* yrow = Y + (size_t)idx * HDC;
  float acc = pb[0];
#pragma unroll 8
  for (int e = 0; e < HDC; ++e) acc += yrow[e] * pw[e];
  out[idx] = acc;
  out[RR*4 + idx] = 0.f;                               // log-det diag == 0
}

extern "C" void kernel_launch(void* const* d_in, const int* in_sizes, int n_in,
                              void* d_out, int out_size, void* d_ws, size_t ws_size,
                              hipStream_t stream) {
  (void)in_sizes; (void)n_in; (void)out_size; (void)ws_size;
  // d_in order: t, x, mask, q[w1,b1,w2,b2,w3,b3], k[...], v[...], proj[w,b]
  const float* x   = (const float*)d_in[1];
  const float* qw1 = (const float*)d_in[3];  const float* qb1 = (const float*)d_in[4];
  const float* qw2 = (const float*)d_in[5];  const float* qb2 = (const float*)d_in[6];
  const float* qw3 = (const float*)d_in[7];  const float* qb3 = (const float*)d_in[8];
  const float* kw1 = (const float*)d_in[9];  const float* kb1 = (const float*)d_in[10];
  const float* kw2 = (const float*)d_in[11]; const float* kb2 = (const float*)d_in[12];
  const float* kw3 = (const float*)d_in[13]; const float* kb3 = (const float*)d_in[14];
  const float* vw1 = (const float*)d_in[15]; const float* vb1 = (const float*)d_in[16];
  const float* vw2 = (const float*)d_in[17]; const float* vb2 = (const float*)d_in[18];
  const float* vw3 = (const float*)d_in[19]; const float* vb3 = (const float*)d_in[20];
  const float* pw  = (const float*)d_in[21]; const float* pb  = (const float*)d_in[22];
  float* out = (float*)d_out;

  char* ws = (char*)d_ws;
  size_t off = 0;
  auto alloc = [&](size_t bytes) {
    char* p = ws + off;
    off += (bytes + 255) & ~(size_t)255;
    return p;
  };
  _Float16* h1q = (_Float16*)alloc((size_t)RR*256*2);
  _Float16* h1k = (_Float16*)alloc((size_t)RR*256*2);
  _Float16* h1v = (_Float16*)alloc((size_t)RR*256*2);
  _Float16* h2q = (_Float16*)alloc((size_t)RR*256*2);
  _Float16* h2k = (_Float16*)alloc((size_t)RR*256*2);
  _Float16* h2v = (_Float16*)alloc((size_t)RR*256*2);
  _Float16* qf  = (_Float16*)alloc((size_t)RR*512*2);
  _Float16* kf  = (_Float16*)alloc((size_t)RR*128*2);
  _Float16* vt  = (_Float16*)alloc((size_t)NB*128*NN*2);
  float*    y   = (float*)   alloc((size_t)RR*512*4);
  _Float16* w2q = (_Float16*)alloc(256*256*2);
  _Float16* w2k = (_Float16*)alloc(256*256*2);
  _Float16* w2v = (_Float16*)alloc(256*256*2);
  _Float16* w3q = (_Float16*)alloc(512*256*2);
  _Float16* w3k = (_Float16*)alloc(128*256*2);
  _Float16* w3v = (_Float16*)alloc(128*256*2);

  // 1) Weight conversion (masks folded in).
  convert_w_kernel<<<256, 256, 0, stream>>>(qw2, w2q, 256, 256, 1);
  convert_w_kernel<<<256, 256, 0, stream>>>(kw2, w2k, 256, 256, 0);
  convert_w_kernel<<<256, 256, 0, stream>>>(vw2, w2v, 256, 256, 0);
  convert_w_kernel<<<512, 256, 0, stream>>>(qw3, w3q, 512, 256, 2);
  convert_w_kernel<<<128, 256, 0, stream>>>(kw3, w3k, 128, 256, 0);
  convert_w_kernel<<<128, 256, 0, stream>>>(vw3, w3v, 128, 256, 0);

  // 2) Layer 1 (K=4, scalar).
  layer1_kernel<<<RR, 256, 0, stream>>>(x, qw1, qb1, kw1, kb1, vw1, vb1, h1q, h1k, h1v);

  // 3) Hidden layer 2: 512 row-panels x (256/128) blocks.
  gemm_k256_kernel<<<1024, 256, 0, stream>>>(h1q, w2q, qb2, h2q, 256, 0, 0);  // relu (MADE)
  gemm_k256_kernel<<<1024, 256, 0, stream>>>(h1k, w2k, kb2, h2k, 256, 1, 0);  // tanh
  gemm_k256_kernel<<<1024, 256, 0, stream>>>(h1v, w2v, vb2, h2v, 256, 1, 0);  // tanh

  // 4) Output layer 3: q -> [RR,512]; k -> [RR,128]; v -> transposed [B*128,N].
  gemm_k256_kernel<<<2048, 256, 0, stream>>>(h2q, w3q, qb3, qf, 512, 2, 0);
  gemm_k256_kernel<<<512,  256, 0, stream>>>(h2k, w3k, kb3, kf, 128, 2, 0);
  gemm_k256_kernel<<<512,  256, 0, stream>>>(h2v, w3v, vb3, vt, 128, 2, 1);

  // 5) Flash attention: 8192 (b,d,h,qtile) waves / 8 per block.
  attention_kernel<<<1024, 256, 0, stream>>>(qf, kf, vt, y);

  // 6) Projection + zero log-det.
  proj_kernel<<<(RR*4)/256, 256, 0, stream>>>(y, pw, pb, out);
}
